// QConv1D_23819888624283
// MI455X (gfx1250) — compile-verified, matrix operand order checked by reading, not run on hardware
//
#include <hip/hip_runtime.h>
#include <hip/hip_bf16.h>

typedef __attribute__((ext_vector_type(16))) _Float16 v16h;
typedef __attribute__((ext_vector_type(8)))  _Float16 v8h;
typedef __attribute__((ext_vector_type(4)))  _Float16 v4h;
typedef __attribute__((ext_vector_type(8)))  float    v8f;

typedef int b128_t __attribute__((vector_size(16)));   // 4 x i32 payload type

#define M_ROWS 32768   // B*S = 8*4096
#define K_DIM  1024
#define N_DIM  1024
#define EPS    0.01f
#define QMAXF  128.0f

#if __has_builtin(__builtin_amdgcn_global_load_async_to_lds_b128) && \
    __has_builtin(__builtin_amdgcn_s_wait_asynccnt)
#define USE_ASYNC_LDS 1
#else
#define USE_ASYNC_LDS 0
#endif

// builtin param types (from hipcc diagnostic): non-const b128_t* in AS1 / AS3
#define GPTR(p) ((__attribute__((address_space(1))) b128_t*)(p))
#define LPTR(p) ((__attribute__((address_space(3))) b128_t*)(p))

// ---------------------------------------------------------------------------
// helpers
// ---------------------------------------------------------------------------
__device__ __forceinline__ float pow2ceil(float m) {
  // m > 0 (>= EPS). exp2(ceil(log2(m))): if mantissa==0 keep, else bump exponent.
  unsigned u    = __float_as_uint(m);
  unsigned mant = u & 0x7FFFFFu;
  unsigned expb = u & 0x7F800000u;
  return (mant == 0u) ? m : __uint_as_float(expb + 0x00800000u);
}

__device__ __forceinline__ v8f wmma_f16(v16h a, v16h b, v8f c) {
  return __builtin_amdgcn_wmma_f32_16x16x32_f16(
      /*neg_a=*/false, a, /*neg_b=*/false, b,
      /*c_mod=*/(short)0, c, /*reuse_a=*/false, /*reuse_b=*/false);
}

// ---------------------------------------------------------------------------
// 1) init column-max buffers to bits(EPS)
// ---------------------------------------------------------------------------
__global__ __launch_bounds__(256) void init_colmax_kernel(unsigned* __restrict__ cx,
                                                          unsigned* __restrict__ cw) {
  int i = blockIdx.x * 256 + threadIdx.x;           // 0..2047
  unsigned e = __float_as_uint(EPS);
  if (i < K_DIM)            cx[i]          = e;     // x column max (per k)
  else if (i < 2 * K_DIM)   cw[i - K_DIM]  = e;     // w column max (per n)
}

// ---------------------------------------------------------------------------
// 2) column abs-max via monotonic uint atomicMax on float bits
// ---------------------------------------------------------------------------
__global__ __launch_bounds__(256) void colmax_kernel(const float* __restrict__ X,
                                                     unsigned* __restrict__ colmax,
                                                     int rowsPerBlock) {
  int col = blockIdx.x * 256 + threadIdx.x;         // 1024 cols total
  int r0  = blockIdx.y * rowsPerBlock;
  float m = 0.0f;
  for (int r = 0; r < rowsPerBlock; ++r) {
    m = fmaxf(m, fabsf(X[(size_t)(r0 + r) * K_DIM + col]));
  }
  atomicMax(colmax + col, __float_as_uint(m));      // |x| bits are order-preserving
}

// ---------------------------------------------------------------------------
// 3) colmax bits -> power-of-2 scale
// ---------------------------------------------------------------------------
__global__ __launch_bounds__(256) void scol_kernel(const unsigned* __restrict__ cx,
                                                   const unsigned* __restrict__ cw,
                                                   float* __restrict__ s_col_x,
                                                   float* __restrict__ s_col_w) {
  int i = blockIdx.x * 256 + threadIdx.x;           // 0..2047
  if (i < K_DIM) {
    s_col_x[i] = pow2ceil(fmaxf(__uint_as_float(cx[i]), EPS));
  } else if (i < 2 * K_DIM) {
    int j = i - K_DIM;
    s_col_w[j] = pow2ceil(fmaxf(__uint_as_float(cw[j]), EPS));
  }
}

// ---------------------------------------------------------------------------
// 4a) quantize x rows: A[i,k] = q * s_col_x[k] (exact f16), alpha[i] = s_row/128
// ---------------------------------------------------------------------------
__global__ __launch_bounds__(256) void quant_x_kernel(const float* __restrict__ x,
                                                      const float* __restrict__ s_col,
                                                      _Float16* __restrict__ Aq,
                                                      float* __restrict__ alpha) {
  __shared__ float red[256];
  const int row = blockIdx.x;
  const int t   = threadIdx.x;
  const float4 xv = ((const float4*)(x + (size_t)row * K_DIM))[t];
  const float4 sv = ((const float4*)s_col)[t];
  float v0 = xv.x / sv.x, v1 = xv.y / sv.y, v2 = xv.z / sv.z, v3 = xv.w / sv.w;
  float mx = fmaxf(fmaxf(fabsf(v0), fabsf(v1)), fmaxf(fabsf(v2), fabsf(v3)));
  red[t] = mx;
  __syncthreads();
  #pragma unroll
  for (int s = 128; s > 0; s >>= 1) {
    if (t < s) red[t] = fmaxf(red[t], red[t + s]);
    __syncthreads();
  }
  const float sr = pow2ceil(fmaxf(red[0], EPS));
  const float qs = QMAXF / sr;
  float q0 = fminf(fmaxf(rintf(v0 * qs), -QMAXF), QMAXF - 1.0f);
  float q1 = fminf(fmaxf(rintf(v1 * qs), -QMAXF), QMAXF - 1.0f);
  float q2 = fminf(fmaxf(rintf(v2 * qs), -QMAXF), QMAXF - 1.0f);
  float q3 = fminf(fmaxf(rintf(v3 * qs), -QMAXF), QMAXF - 1.0f);
  v4h a;
  a[0] = (_Float16)(q0 * sv.x);
  a[1] = (_Float16)(q1 * sv.y);
  a[2] = (_Float16)(q2 * sv.z);
  a[3] = (_Float16)(q3 * sv.w);
  ((v4h*)(Aq + (size_t)row * K_DIM))[t] = a;        // 8B coalesced store
  if (t == 0) alpha[row] = sr * (1.0f / QMAXF);
}

// ---------------------------------------------------------------------------
// 4b) quantize weight rows (k), store TRANSPOSED: Bt[n,k] = q * s_row_w[k]
// ---------------------------------------------------------------------------
__global__ __launch_bounds__(256) void quant_w_kernel(const float* __restrict__ w,
                                                      const float* __restrict__ s_col_w,
                                                      _Float16* __restrict__ Bt) {
  __shared__ float red[256];
  const int krow = blockIdx.x;                      // 1024 rows of weight
  const int t    = threadIdx.x;
  const float4 wv = ((const float4*)(w + (size_t)krow * N_DIM))[t];
  const float4 sv = ((const float4*)s_col_w)[t];
  float v0 = wv.x / sv.x, v1 = wv.y / sv.y, v2 = wv.z / sv.z, v3 = wv.w / sv.w;
  float mx = fmaxf(fmaxf(fabsf(v0), fabsf(v1)), fmaxf(fabsf(v2), fabsf(v3)));
  red[t] = mx;
  __syncthreads();
  #pragma unroll
  for (int s = 128; s > 0; s >>= 1) {
    if (t < s) red[t] = fmaxf(red[t], red[t + s]);
    __syncthreads();
  }
  const float sr = pow2ceil(fmaxf(red[0], EPS));
  const float qs = QMAXF / sr;
  float q0 = fminf(fmaxf(rintf(v0 * qs), -QMAXF), QMAXF - 1.0f);
  float q1 = fminf(fmaxf(rintf(v1 * qs), -QMAXF), QMAXF - 1.0f);
  float q2 = fminf(fmaxf(rintf(v2 * qs), -QMAXF), QMAXF - 1.0f);
  float q3 = fminf(fmaxf(rintf(v3 * qs), -QMAXF), QMAXF - 1.0f);
  const int n = t * 4;
  Bt[(size_t)(n + 0) * K_DIM + krow] = (_Float16)(q0 * sr);
  Bt[(size_t)(n + 1) * K_DIM + krow] = (_Float16)(q1 * sr);
  Bt[(size_t)(n + 2) * K_DIM + krow] = (_Float16)(q2 * sr);
  Bt[(size_t)(n + 3) * K_DIM + krow] = (_Float16)(q3 * sr);
}

// ---------------------------------------------------------------------------
// 5) WMMA GEMM, double-buffered async-to-LDS pipeline.
// 128x128 tile per block, 8 waves of 32M x 64N, K staged in blocks of 32.
// ---------------------------------------------------------------------------
#define KSTAGE     32
#define LDS_STRIDE 40   // 32 halves + 8 pad -> 80B row stride, 16B aligned chunks

__device__ __forceinline__ v16h frag_ld(const _Float16* lds, int row, int kbase) {
  const v8h lo = *(const v8h*)(lds + row * LDS_STRIDE + kbase);
  const v8h hi = *(const v8h*)(lds + row * LDS_STRIDE + kbase + 16);
  return __builtin_shufflevector(lo, hi, 0,1,2,3,4,5,6,7,8,9,10,11,12,13,14,15);
}

__global__ __launch_bounds__(256) void gemm_kernel(const _Float16* __restrict__ Aq,
                                                   const _Float16* __restrict__ Bt,
                                                   const float* __restrict__ alpha,
                                                   const float* __restrict__ s_col_w,
                                                   const float* __restrict__ bias,
                                                   float* __restrict__ out) {
  __shared__ _Float16 As[2][128 * LDS_STRIDE];
  __shared__ _Float16 Bs[2][128 * LDS_STRIDE];
  const int tid  = threadIdx.x;
  const int lane = tid & 31;
  const int wave = tid >> 5;
  const int wm   = (wave & 3) * 32;                 // wave's M offset in tile
  const int wn   = (wave >> 2) * 64;                // wave's N offset in tile
  const int mBase = blockIdx.y * 128;
  const int nBase = blockIdx.x * 128;
  const int lrow  = lane & 15;
  const int lks   = (lane >> 4) * 8;                // K sub-offset per lane half

  // staging geometry: 128 rows x 32 halves = 512 x 16B chunks per matrix
  const int c0row = tid >> 2;                       // chunk row for j=0 (cidx=tid)
  const int c0kc  = (tid & 3) * 8;                  // chunk K offset (halves)
  const int c1row = (tid + 256) >> 2;               // chunk row for j=1
  const int c1kc  = c0kc;

  v8f acc[2][4];
  #pragma unroll
  for (int i = 0; i < 2; ++i)
    #pragma unroll
    for (int j = 0; j < 4; ++j)
      acc[i][j] = (v8f){0.f, 0.f, 0.f, 0.f, 0.f, 0.f, 0.f, 0.f};

#if USE_ASYNC_LDS
  // ---- async double-buffered pipeline: one barrier per stage ----
  {
    // prologue: stage 0 -> buffer 0
    __builtin_amdgcn_global_load_async_to_lds_b128(
        GPTR(Aq + (size_t)(mBase + c0row) * K_DIM + c0kc), LPTR(&As[0][c0row * LDS_STRIDE + c0kc]), 0, 0);
    __builtin_amdgcn_global_load_async_to_lds_b128(
        GPTR(Aq + (size_t)(mBase + c1row) * K_DIM + c1kc), LPTR(&As[0][c1row * LDS_STRIDE + c1kc]), 0, 0);
    __builtin_amdgcn_global_load_async_to_lds_b128(
        GPTR(Bt + (size_t)(nBase + c0row) * K_DIM + c0kc), LPTR(&Bs[0][c0row * LDS_STRIDE + c0kc]), 0, 0);
    __builtin_amdgcn_global_load_async_to_lds_b128(
        GPTR(Bt + (size_t)(nBase + c1row) * K_DIM + c1kc), LPTR(&Bs[0][c1row * LDS_STRIDE + c1kc]), 0, 0);
  }
  for (int s = 0; s < K_DIM / KSTAGE; ++s) {
    const int p = s & 1;
    __builtin_amdgcn_s_wait_asynccnt(0);            // my stage-s chunks landed in LDS
    __syncthreads();                                // all waves' chunks landed; buf 1-p free
    if (s + 1 < K_DIM / KSTAGE) {                   // overlap stage s+1 loads with compute
      const int kb = (s + 1) * KSTAGE;
      __builtin_amdgcn_global_load_async_to_lds_b128(
          GPTR(Aq + (size_t)(mBase + c0row) * K_DIM + kb + c0kc), LPTR(&As[1 - p][c0row * LDS_STRIDE + c0kc]), 0, 0);
      __builtin_amdgcn_global_load_async_to_lds_b128(
          GPTR(Aq + (size_t)(mBase + c1row) * K_DIM + kb + c1kc), LPTR(&As[1 - p][c1row * LDS_STRIDE + c1kc]), 0, 0);
      __builtin_amdgcn_global_load_async_to_lds_b128(
          GPTR(Bt + (size_t)(nBase + c0row) * K_DIM + kb + c0kc), LPTR(&Bs[1 - p][c0row * LDS_STRIDE + c0kc]), 0, 0);
      __builtin_amdgcn_global_load_async_to_lds_b128(
          GPTR(Bt + (size_t)(nBase + c1row) * K_DIM + kb + c1kc), LPTR(&Bs[1 - p][c1row * LDS_STRIDE + c1kc]), 0, 0);
    }
    const v16h a0 = frag_ld(As[p], wm +  0 + lrow, lks);
    const v16h a1 = frag_ld(As[p], wm + 16 + lrow, lks);
    const v16h b0 = frag_ld(Bs[p], wn +  0 + lrow, lks);
    const v16h b1 = frag_ld(Bs[p], wn + 16 + lrow, lks);
    const v16h b2 = frag_ld(Bs[p], wn + 32 + lrow, lks);
    const v16h b3 = frag_ld(Bs[p], wn + 48 + lrow, lks);
    acc[0][0] = wmma_f16(a0, b0, acc[0][0]);
    acc[0][1] = wmma_f16(a0, b1, acc[0][1]);
    acc[0][2] = wmma_f16(a0, b2, acc[0][2]);
    acc[0][3] = wmma_f16(a0, b3, acc[0][3]);
    acc[1][0] = wmma_f16(a1, b0, acc[1][0]);
    acc[1][1] = wmma_f16(a1, b1, acc[1][1]);
    acc[1][2] = wmma_f16(a1, b2, acc[1][2]);
    acc[1][3] = wmma_f16(a1, b3, acc[1][3]);
  }
#else
  // ---- fallback: synchronous staging through VGPRs ----
  for (int s = 0; s < K_DIM / KSTAGE; ++s) {
    const int kb = s * KSTAGE;
    __syncthreads();
    *(uint4*)(&As[0][c0row * LDS_STRIDE + c0kc]) =
        *(const uint4*)(Aq + (size_t)(mBase + c0row) * K_DIM + kb + c0kc);
    *(uint4*)(&As[0][c1row * LDS_STRIDE + c1kc]) =
        *(const uint4*)(Aq + (size_t)(mBase + c1row) * K_DIM + kb + c1kc);
    *(uint4*)(&Bs[0][c0row * LDS_STRIDE + c0kc]) =
        *(const uint4*)(Bt + (size_t)(nBase + c0row) * K_DIM + kb + c0kc);
    *(uint4*)(&Bs[0][c1row * LDS_STRIDE + c1kc]) =
        *(const uint4*)(Bt + (size_t)(nBase + c1row) * K_DIM + kb + c1kc);
    __syncthreads();
    const v16h a0 = frag_ld(As[0], wm +  0 + lrow, lks);
    const v16h a1 = frag_ld(As[0], wm + 16 + lrow, lks);
    const v16h b0 = frag_ld(Bs[0], wn +  0 + lrow, lks);
    const v16h b1 = frag_ld(Bs[0], wn + 16 + lrow, lks);
    const v16h b2 = frag_ld(Bs[0], wn + 32 + lrow, lks);
    const v16h b3 = frag_ld(Bs[0], wn + 48 + lrow, lks);
    acc[0][0] = wmma_f16(a0, b0, acc[0][0]);
    acc[0][1] = wmma_f16(a0, b1, acc[0][1]);
    acc[0][2] = wmma_f16(a0, b2, acc[0][2]);
    acc[0][3] = wmma_f16(a0, b3, acc[0][3]);
    acc[1][0] = wmma_f16(a1, b0, acc[1][0]);
    acc[1][1] = wmma_f16(a1, b1, acc[1][1]);
    acc[1][2] = wmma_f16(a1, b2, acc[1][2]);
    acc[1][3] = wmma_f16(a1, b3, acc[1][3]);
  }
#endif

  // epilogue: C/D layout => lane 0-15: N=lane, M=r; lane 16-31: N=lane-16, M=r+8
  const int cm = (lane >> 4) * 8;
  const int cn = lane & 15;
  #pragma unroll
  for (int ni = 0; ni < 4; ++ni) {
    const int ncol  = nBase + wn + ni * 16 + cn;
    const float bet = s_col_w[ncol] * (1.0f / QMAXF);
    const float bz  = bias[ncol];
    #pragma unroll
    for (int mi = 0; mi < 2; ++mi) {
      #pragma unroll
      for (int r = 0; r < 8; ++r) {
        const int mrow = mBase + wm + mi * 16 + cm + r;
        out[(size_t)mrow * N_DIM + ncol] = acc[mi][ni][r] * alpha[mrow] * bet + bz;
      }
    }
  }
}

// ---------------------------------------------------------------------------
// host launcher
// ---------------------------------------------------------------------------
extern "C" void kernel_launch(void* const* d_in, const int* in_sizes, int n_in,
                              void* d_out, int out_size, void* d_ws, size_t ws_size,
                              hipStream_t stream) {
  const float* x    = (const float*)d_in[0];   // (8,4096,1024)
  const float* w    = (const float*)d_in[1];   // (1024,1024)
  const float* bias = (const float*)d_in[2];   // (1024,)
  float* out = (float*)d_out;

  char* ws = (char*)d_ws;
  unsigned* colmax_x = (unsigned*)(ws + 0);          // 4KB
  float*    s_col_x  = (float*)   (ws + 4096);       // 4KB
  unsigned* colmax_w = (unsigned*)(ws + 8192);       // 4KB
  float*    s_col_w  = (float*)   (ws + 12288);      // 4KB
  float*    alpha    = (float*)   (ws + 16384);      // 128KB
  _Float16* Btq      = (_Float16*)(ws + 147456);     // 2MB  (N-major, stride K)
  _Float16* Aq       = (_Float16*)(ws + 2244608);    // 64MB (row-major, stride K)

  init_colmax_kernel<<<8, 256, 0, stream>>>(colmax_x, colmax_w);
  colmax_kernel<<<dim3(K_DIM / 256, M_ROWS / 128), 256, 0, stream>>>(x, colmax_x, 128);
  colmax_kernel<<<dim3(N_DIM / 256, K_DIM / 128), 256, 0, stream>>>(w, colmax_w, 128);
  scol_kernel<<<8, 256, 0, stream>>>(colmax_x, colmax_w, s_col_x, s_col_w);
  quant_x_kernel<<<M_ROWS, 256, 0, stream>>>(x, s_col_x, Aq, alpha);
  quant_w_kernel<<<K_DIM, 256, 0, stream>>>(w, s_col_w, Btq);
  gemm_kernel<<<dim3(N_DIM / 128, M_ROWS / 128), 256, 0, stream>>>(
      Aq, Btq, alpha, s_col_w, bias, out);
}
